// SNN_61246233641407
// MI455X (gfx1250) — matrix operand order repaired and checked by reading, not compiled, not run on hardware
//
#include <hip/hip_runtime.h>

// ---------------- Problem constants (match reference) ----------------
static constexpr int    Dd  = 4096;   // feature dim
static constexpr int    Bb  = 256;    // batch
static constexpr int    NST = 30;     // relaxation steps
#define DT 0.5f

typedef __bf16 bf16;
typedef __attribute__((ext_vector_type(16))) __bf16 v16bf;
typedef __attribute__((ext_vector_type(8)))  __bf16 v8bf;
typedef __attribute__((ext_vector_type(8)))  float  v8f;

// ---------------- helpers ----------------
__device__ __forceinline__ unsigned short f2bf(float f) {
  unsigned u = __builtin_bit_cast(unsigned, f);
  u += 0x7FFFu + ((u >> 16) & 1u);              // round-to-nearest-even
  return (unsigned short)(u >> 16);
}

__device__ __forceinline__ void store4bf(bf16* dst, float a, float b, float c, float d) {
  unsigned lo = (unsigned)f2bf(a) | ((unsigned)f2bf(b) << 16);
  unsigned hi = (unsigned)f2bf(c) | ((unsigned)f2bf(d) << 16);
  uint2 v; v.x = lo; v.y = hi;
  *(uint2*)dst = v;
}

__device__ __forceinline__ float clip01(float x) {
  return fminf(fmaxf(x, 0.0f), 1.0f);
}

// A fragment: 16x32 bf16 activations. contiguous k order (k = 16h + s, same as B):
// lane (m = lane&15, h = lane>>4): elems[0..7]  = A[m][k0 + h*16 .. +7]
//                                  elems[8..15] = A[m][k0 + h*16 + 8 .. +7]
__device__ __forceinline__ v16bf load_a_frag(const bf16* A, int m0, int k0, int lane) {
  int m = lane & 15, h = lane >> 4;
  const bf16* p = A + (size_t)(m0 + m) * Dd + k0 + h * 16;
  union { v16bf v; v8bf half[2]; } u;
  u.half[0] = *(const v8bf*)(p);
  u.half[1] = *(const v8bf*)(p + 8);
  return u.v;
}

// B fragment: 32x16 bf16 with B[k][n] = W[n][k]  (@W.T => contiguous row
// segment of W per lane; single 32B load).
__device__ __forceinline__ v16bf load_b_frag(const bf16* W, int n0, int k0, int lane) {
  int n = lane & 15, h = lane >> 4;
  return *(const v16bf*)(W + (size_t)(n0 + n) * Dd + k0 + h * 16);
}

// 64M x 32N per-wave tile: 8 accumulator frags, 8 wmma per 12 loads per K-step.
// Weights shared across 4 M-frags in-register (4x less L2 weight traffic than
// a 16M tile would need).
__device__ __forceinline__ void gemm_acc8(v8f c[4][2], const bf16* A, const bf16* W,
                                          int m0, int n0, int lane) {
#pragma unroll 2
  for (int k = 0; k < Dd; k += 32) {
    v16bf a[4];
#pragma unroll
    for (int i = 0; i < 4; ++i) a[i] = load_a_frag(A, m0 + 16 * i, k, lane);
    v16bf b0 = load_b_frag(W, n0,      k, lane);
    v16bf b1 = load_b_frag(W, n0 + 16, k, lane);
#pragma unroll
    for (int i = 0; i < 4; ++i) {
      c[i][0] = __builtin_amdgcn_wmma_f32_16x16x32_bf16(false, a[i], false, b0, (short)0, c[i][0], false, false);
      c[i][1] = __builtin_amdgcn_wmma_f32_16x16x32_bf16(false, a[i], false, b1, (short)0, c[i][1], false, false);
    }
  }
}

// ---------------- conversion kernels ----------------
// fp32 -> bf16 for the 5 weight matrices (one-time; afterwards the four
// per-step matrices are 4*32MB bf16 and live in the 192MB L2).
__global__ __launch_bounds__(256) void w_to_bf16_k(
    const float* w0, const float* w1, const float* w2, const float* w3,
    const float* w4, bf16* dst_base) {
  const float* srcs[5] = {w0, w1, w2, w3, w4};
  const float* src = srcs[blockIdx.z];
  bf16* dst = dst_base + (size_t)blockIdx.z * Dd * Dd;
  size_t i = ((size_t)blockIdx.x * blockDim.x + threadIdx.x) * 4;
  float4 v = *(const float4*)(src + i);
  store4bf(dst + i, v.x, v.y, v.z, v.w);
}

// rho(x)=clip(x,0,1) then cast bf16; flat over 4*gridDim.x*256 floats.
__global__ __launch_bounds__(256) void rho_bf16_k(const float* s, bf16* r) {
  size_t i = ((size_t)blockIdx.x * blockDim.x + threadIdx.x) * 4;
  float4 v = *(const float4*)(s + i);
  store4bf(r + i, clip01(v.x), clip01(v.y), clip01(v.z), clip01(v.w));
}

// ---------------- GEMM: top_in = rho(s3) @ W4^T + b4 (fp32 out) ----------------
// Block: 4 waves stacked in M (4 x 64 = 256 rows) x 32N -> grid.x = D/32.
__global__ __launch_bounds__(128) void topin_gemm_k(const bf16* A, const bf16* W,
                                                    const float* bias, float* out) {
  int lane = threadIdx.x & 31;
  int m0 = (threadIdx.x >> 5) * 64;
  int n0 = blockIdx.x * 32;
  v8f z = {0.f,0.f,0.f,0.f,0.f,0.f,0.f,0.f};
  v8f c[4][2] = {{z,z},{z,z},{z,z},{z,z}};
  gemm_acc8(c, A, W, m0, n0, lane);
  int nc = lane & 15, hh = lane >> 4;
  for (int fm = 0; fm < 4; fm++)
    for (int fn = 0; fn < 2; fn++) {
      int col = n0 + fn * 16 + nc;
      float bv = bias[col];
      for (int v = 0; v < 8; v++) {
        int row = m0 + fm * 16 + v + 8 * hh;
        out[(size_t)row * Dd + col] = c[fm][fn][v] + bv;
      }
    }
}

// ---------------- fused per-step kernel: 3 jobs (layers), 4 GEMMs ----------------
// z = 0 (long pole, dispatched first for LPT packing):
//       S1 <- clip(S1 + DT*(-S1 + r2@W2^T + r0@W1^T + b2)) ; rn1 <- bf16(S1')
// z = 1: S0 <- clip(S0 + DT*(-S0 + r1@W0^T + b0))          ; rn0 <- bf16(S0')
// z = 2: S2 <- clip(S2 + DT*(-S2 + top_in + r1@W3^T))      ; rn2 <- bf16(S2')
// (rho(clip01(x)) == clip01(x), so the epilogue emits next step's activations
//  directly; r buffers ping-pong across steps to avoid RAW across jobs.)
__global__ __launch_bounds__(128) void ep_step_gemm_k(
    const bf16* r0, const bf16* r1, const bf16* r2,
    const bf16* W0b, const bf16* W1b, const bf16* W2b, const bf16* W3b,
    const float* b0, const float* b2, const float* top_in,
    float* S0, float* S1, float* S2,
    bf16* rn0, bf16* rn1, bf16* rn2) {
  int lane = threadIdx.x & 31;
  int m0 = (threadIdx.x >> 5) * 64;
  int n0 = blockIdx.x * 32;
  v8f z = {0.f,0.f,0.f,0.f,0.f,0.f,0.f,0.f};
  v8f c[4][2] = {{z,z},{z,z},{z,z},{z,z}};
  int job = blockIdx.z;

  if (job == 0) {                                  // layer 1 (double GEMM)
    gemm_acc8(c, r2, W2b, m0, n0, lane);
    gemm_acc8(c, r0, W1b, m0, n0, lane);
  } else if (job == 1) {                           // layer 0
    gemm_acc8(c, r1, W0b, m0, n0, lane);
  } else {                                         // layer 2
    gemm_acc8(c, r1, W3b, m0, n0, lane);
  }

  const float* bias = (job == 0) ? b2 : ((job == 1) ? b0 : nullptr);
  float* S  = (job == 0) ? S1  : ((job == 1) ? S0  : S2);
  bf16*  RN = (job == 0) ? rn1 : ((job == 1) ? rn0 : rn2);

  int nc = lane & 15, hh = lane >> 4;
  for (int fm = 0; fm < 4; fm++)
    for (int fn = 0; fn < 2; fn++) {
      int col = n0 + fn * 16 + nc;
      float bv = bias ? bias[col] : 0.0f;
      for (int v = 0; v < 8; v++) {
        int row = m0 + fm * 16 + v + 8 * hh;
        size_t idx = (size_t)row * Dd + col;
        float pre = c[fm][fn][v] + bv;
        if (job == 2) pre += top_in[idx];
        float cold = S[idx];
        float ns = clip01(cold + DT * (pre - cold));   // c + DT*(-c + pre)
        S[idx]  = ns;
        RN[idx] = __builtin_bit_cast(bf16, f2bf(ns));  // next step's rho snapshot
      }
    }
}

// ---------------- host-side orchestration ----------------
extern "C" void kernel_launch(void* const* d_in, const int* in_sizes, int n_in,
                              void* d_out, int out_size, void* d_ws, size_t ws_size,
                              hipStream_t stream) {
  (void)in_sizes; (void)n_in; (void)out_size; (void)ws_size;
  const float* s0 = (const float*)d_in[0];
  const float* s1 = (const float*)d_in[1];
  const float* s2 = (const float*)d_in[2];
  const float* s3 = (const float*)d_in[3];
  const float* W0 = (const float*)d_in[4];
  const float* b0 = (const float*)d_in[5];
  const float* W1 = (const float*)d_in[6];
  const float* W2 = (const float*)d_in[7];
  const float* b2 = (const float*)d_in[8];
  const float* W3 = (const float*)d_in[9];
  const float* W4 = (const float*)d_in[10];
  const float* b4 = (const float*)d_in[11];

  const size_t BD = (size_t)Bb * Dd;           // 1,048,576
  const size_t DD = (size_t)Dd * Dd;           // 16,777,216

  float* out = (float*)d_out;                  // states live in d_out: [s0|s1|s2]
  float* S0 = out;
  float* S1 = out + BD;
  float* S2 = out + 2 * BD;

  // workspace layout (bytes): 5*DD*2 weights | RA 3*BD*2 | RB 3*BD*2 | r3 BD*2 | top BD*4
  char* ws    = (char*)d_ws;
  bf16* Wb    = (bf16*)ws;                                   // W0..W4 bf16
  bf16* RA    = (bf16*)(ws + 5 * DD * 2);                    // rho ping
  bf16* RB    = (bf16*)(ws + 5 * DD * 2 + 3 * BD * 2);       // rho pong
  bf16* r3    = (bf16*)(ws + 5 * DD * 2 + 6 * BD * 2);       // rho(s3)
  float* top  = (float*)(ws + 5 * DD * 2 + 7 * BD * 2);      // top_in fp32

  // 1) states start from the inputs
  hipMemcpyAsync(S0, s0, BD * sizeof(float), hipMemcpyDeviceToDevice, stream);
  hipMemcpyAsync(S1, s1, BD * sizeof(float), hipMemcpyDeviceToDevice, stream);
  hipMemcpyAsync(S2, s2, BD * sizeof(float), hipMemcpyDeviceToDevice, stream);

  // 2) one-time fp32 -> bf16 weight conversion
  w_to_bf16_k<<<dim3((unsigned)(DD / 4 / 256), 1, 5), 256, 0, stream>>>(W0, W1, W2, W3, W4, Wb);

  // 3) constant top_in = rho(s3) @ W4^T + b4
  rho_bf16_k<<<dim3((unsigned)(BD / 4 / 256)), 256, 0, stream>>>(s3, r3);
  topin_gemm_k<<<dim3(Dd / 32), 128, 0, stream>>>(r3, Wb + 4 * DD, b4, top);

  // 4) initial rho snapshot of the (contiguous) states -> RA
  rho_bf16_k<<<dim3((unsigned)(3 * BD / 4 / 256)), 256, 0, stream>>>(out, RA);

  // 5) 30 sequential relaxation steps; rho fused into GEMM epilogue with
  //    ping-pong activation buffers. 384 x 4-wave blocks per step; the
  //    double-GEMM layer-1 job sits at z=0 so its blocks launch first.
  for (int t = 0; t < NST; ++t) {
    bf16* rc = (t & 1) ? RB : RA;   // current activations
    bf16* rn = (t & 1) ? RA : RB;   // next activations (written by epilogue)
    ep_step_gemm_k<<<dim3(Dd / 32, 1, 3), 128, 0, stream>>>(
        rc, rc + BD, rc + 2 * BD,
        Wb, Wb + DD, Wb + 2 * DD, Wb + 3 * DD,
        b0, b2, top, S0, S1, S2,
        rn, rn + BD, rn + 2 * BD);
  }
  // d_out already holds [s0; s1; s2] stacked -> nothing to copy.
}
